// MolecularGCN_3478923510589
// MI455X (gfx1250) — matrix operand order repaired and checked by reading, not compiled, not run on hardware
//
#include <hip/hip_runtime.h>
#include <hip/hip_bf16.h>
#include <stdint.h>

#define N_NODES 524288
#define N_EDGES 2097152
#define F_IN 75
#define D 128
#define KPAD 96   // 75 padded up to 3 x 32 for bf16 WMMA K-steps

typedef __attribute__((ext_vector_type(16))) __bf16 v16bf;
typedef __attribute__((ext_vector_type(8)))  float  v8f;
typedef __attribute__((ext_vector_type(4)))  unsigned int v4u;
typedef __attribute__((ext_vector_type(8)))  int    v8i;
typedef __attribute__((ext_vector_type(4)))  int    v4i;

__device__ __forceinline__ unsigned short f32_to_bf16(float f) {
  unsigned int u = __float_as_uint(f);
  u += 0x7FFFu + ((u >> 16) & 1u);   // round-to-nearest-even
  return (unsigned short)(u >> 16);
}

// ---------------------------------------------------------------------------
// Tensor Data Mover: 2D tile (rows x cols, bf16) global -> LDS.
// D# built per CDNA5 ISA 8.3/8.4: group0 {count=1, lds_addr, global_addr,
// type=2}; group1 {data_size=2B, tensor_dim0/1 = tile dims, tile_dim0/1,
// tensor_dim0_stride}. Groups 2/3 zero (2D tensor).
// clang-23 toolchain: 6-arg builtin (g0, g1, g2, g3, g4, cpol).
// ---------------------------------------------------------------------------
__device__ __forceinline__ void tdm_load_tile_2d(unsigned int lds_byte_addr,
                                                 const void* gptr,
                                                 unsigned int cols,
                                                 unsigned int rows,
                                                 unsigned int stride_elems) {
  unsigned long long ga = (unsigned long long)(uintptr_t)gptr;
  v4u g0;
  g0[0] = 1u;                                        // count=1, user mode
  g0[1] = lds_byte_addr;                             // lds_addr[31:0]
  g0[2] = (unsigned int)ga;                          // global_addr[31:0]
  g0[3] = (unsigned int)(ga >> 32) | (2u << 30);     // addr[56:32] | type=2
  v8i g1;
  g1[0] = (int)(1u << 16);                           // data_size = 1 -> 2 bytes
  g1[1] = (int)((cols & 0xFFFFu) << 16);             // tensor_dim0 lo16 @63:48
  g1[2] = (int)(((cols >> 16) & 0xFFFFu) |           // tensor_dim0 hi16
                ((rows & 0xFFFFu) << 16));           // tensor_dim1 lo16
  g1[3] = (int)(((rows >> 16) & 0xFFFFu) |           // tensor_dim1 hi16
                ((cols & 0xFFFFu) << 16));           // tile_dim0
  g1[4] = (int)(rows & 0xFFFFu);                     // tile_dim1 (tile_dim2=0)
  g1[5] = (int)stride_elems;                         // tensor_dim0_stride lo32
  g1[6] = 0;
  g1[7] = 0;
  v4i g2 = {0, 0, 0, 0};
  v4i g3 = {0, 0, 0, 0};
  v8i g4 = {0, 0, 0, 0, 0, 0, 0, 0};
  __builtin_amdgcn_tensor_load_to_lds(g0, g1, g2, g3, g4, 0);
}

// ---------------------------------------------------------------------------
// WMMA fragment loaders (wave32, contiguous-K order shared by A and B, bf16 16x16x32).
// A 16x32: lane l holds row l&15, elems 0..15 = 16 contiguous K starting
//          at K + (l>>4)*16.
// B 32x16 from transposed-weight image WT[n][k]: lane l holds col n=l&15,
//          elems 0..15 = 16 contiguous K starting at K + (l>>4)*16.
// ---------------------------------------------------------------------------
__device__ __forceinline__ v16bf load_afrag(const unsigned short* p) {
  union { v16bf v; uint4 u[2]; } t;
  t.u[0] = *(const uint4*)(p);        // K+0..7   (or +16..23 in upper half-wave)
  t.u[1] = *(const uint4*)(p + 8);    // K+8..15  (or +24..31)
  return t.v;
}
__device__ __forceinline__ v16bf load_bfrag(const unsigned short* p) {
  union { v16bf v; uint4 u[2]; } t;
  t.u[0] = *(const uint4*)(p);        // 16 contiguous K values (32 bytes)
  t.u[1] = *(const uint4*)(p + 8);
  return t.v;
}
__device__ __forceinline__ v8f wmma_bf16(v16bf a, v16bf b, v8f c) {
  return __builtin_amdgcn_wmma_f32_16x16x32_bf16(false, a, false, b,
                                                 (short)0, c, false, false);
}

// ---------------------------------------------------------------------------
// Weight prep: transpose + fp32->bf16. wt: 6 matrices [n][k] (Wc0..2, Wr0..2),
// wti: init weight [n][96] zero-padded past k=75.
// ---------------------------------------------------------------------------
__global__ void __launch_bounds__(256) gcn_prep_kernel(
    const float* __restrict__ Wc, const float* __restrict__ Wr,
    const float* __restrict__ Wi, unsigned short* __restrict__ wt,
    unsigned short* __restrict__ wti) {
  const int T1 = 6 * D * D;
  const int T2 = D * KPAD;
  int gid = blockIdx.x * 256 + threadIdx.x;
  if (gid < T1) {
    int mat = gid / (D * D);
    int rem = gid - mat * D * D;
    int n = rem / D, k = rem - n * D;
    const float* Ws = (mat < 3) ? (Wc + mat * D * D) : (Wr + (mat - 3) * D * D);
    wt[gid] = f32_to_bf16(Ws[k * D + n]);          // WT[mat][n][k] = W[k][n]
  } else if (gid < T1 + T2) {
    int idx = gid - T1;
    int n = idx / KPAD, k = idx - n * KPAD;
    float v = (k < F_IN) ? Wi[k * D + n] : 0.0f;
    wti[idx] = f32_to_bf16(v);
  }
}

// ---------------------------------------------------------------------------
// h0 = node_feats(f32 [N,75]) @ W_init -> bf16 [N,128]. K padded to 96.
// A fragments hoisted to registers; nt loop fully unrolled.
// ---------------------------------------------------------------------------
__global__ void __launch_bounds__(256) gcn_init_kernel(
    const float* __restrict__ nf, const unsigned short* __restrict__ wti,
    unsigned short* __restrict__ hout) {
  __shared__ unsigned short sA[D * KPAD];
  __shared__ unsigned short sW[D * KPAD];
  const int tid = threadIdx.x;
  const int lane = tid & 31, wave = tid >> 5;
  const int m0 = blockIdx.x * D;

  if (tid < 32)
    tdm_load_tile_2d((unsigned int)(uintptr_t)sW, wti, KPAD, D, KPAD);

  for (int idx = tid; idx < D * KPAD; idx += 256) {
    int r = idx / KPAD, k = idx - r * KPAD;
    float v = (k < F_IN) ? nf[(size_t)(m0 + r) * F_IN + k] : 0.0f;
    sA[idx] = f32_to_bf16(v);
  }
  if (tid < 32) __builtin_amdgcn_s_wait_tensorcnt(0);
  __syncthreads();

  const int row = lane & 15, half = lane >> 4;
  const int kbA = half * 16, kbB = half * 16;
  const unsigned short* a = sA + (wave * 16 + row) * KPAD;

  v16bf af[3];
#pragma unroll
  for (int kk = 0; kk < 3; ++kk) af[kk] = load_afrag(a + kk * 32 + kbA);

#pragma unroll
  for (int nt = 0; nt < 8; ++nt) {
    v8f acc;
#pragma unroll
    for (int r = 0; r < 8; ++r) acc[r] = 0.0f;
    const unsigned short* b = sW + (nt * 16 + row) * KPAD;
#pragma unroll
    for (int kk = 0; kk < 3; ++kk)
      acc = wmma_bf16(af[kk], load_bfrag(b + kk * 32 + kbB), acc);
    const int col = nt * 16 + row;
    const int rbase = m0 + wave * 16 + half * 8;
#pragma unroll
    for (int r = 0; r < 8; ++r)
      hout[(size_t)(rbase + r) * D + col] = f32_to_bf16(acc[r]);
  }
}

// ---------------------------------------------------------------------------
// agg zero + edge scatter-add. agg (bf16 [N,128]) fits in L2 (128MB < 192MB);
// packed bf16 atomics keep the RMW traffic in-cache and halve the op count.
// ---------------------------------------------------------------------------
__global__ void __launch_bounds__(256) gcn_zero_kernel(uint4* __restrict__ p) {
  size_t i = (size_t)blockIdx.x * 256 + threadIdx.x;
  uint4 z; z.x = z.y = z.z = z.w = 0u;
  p[i] = z;
}

__global__ void __launch_bounds__(256) gcn_scatter_kernel(
    const int* __restrict__ src, const int* __restrict__ dst,
    const unsigned short* __restrict__ h, unsigned short* __restrict__ agg) {
  const int gid = blockIdx.x * 256 + threadIdx.x;
  const int e = gid >> 4;          // edge
  const int c = gid & 15;          // 8-element chunk within the 128-wide row
  const int s = src[e];
  const int d = dst[e];
  const uint4 v = *(const uint4*)(h + (size_t)s * D + c * 8);
  unsigned long long a =
      (unsigned long long)(uintptr_t)(agg + (size_t)d * D + c * 8);
  asm volatile("global_atomic_pk_add_bf16 %0, %1, off"
               :: "v"(a), "v"(v.x) : "memory");
  asm volatile("global_atomic_pk_add_bf16 %0, %1, off offset:4"
               :: "v"(a), "v"(v.y) : "memory");
  asm volatile("global_atomic_pk_add_bf16 %0, %1, off offset:8"
               :: "v"(a), "v"(v.z) : "memory");
  asm volatile("global_atomic_pk_add_bf16 %0, %1, off offset:12"
               :: "v"(a), "v"(v.w) : "memory");
}

// ---------------------------------------------------------------------------
// Fused layer: h_next = relu(agg@Wc + bc) + relu(h@Wr + br).
// 128x128 output tile per workgroup, 8 waves x (16 rows x 128 cols) each.
// All four 32KB tiles come in via TDM (128KB dynamic LDS). A fragments for
// both GEMMs are hoisted into registers (8 x v16bf = 64 VGPRs); nt loop is
// fully unrolled so B-tile ds_load_b128s pipeline against the WMMA stream.
// ---------------------------------------------------------------------------
__global__ void __launch_bounds__(256) gcn_layer_kernel(
    const unsigned short* __restrict__ agg, const unsigned short* __restrict__ hin,
    const unsigned short* __restrict__ wtc, const unsigned short* __restrict__ wtr,
    const float* __restrict__ bcl, const float* __restrict__ brl,
    unsigned short* __restrict__ hout, float* __restrict__ fout, int last) {
  extern __shared__ char smem[];
  unsigned short* sXc = (unsigned short*)smem;   // agg tile  [128][128]
  unsigned short* sXr = sXc + D * D;             // h tile
  unsigned short* sWc = sXr + D * D;             // WTc [n][k]
  unsigned short* sWr = sWc + D * D;             // WTr [n][k]

  const int tid = threadIdx.x;
  const int lane = tid & 31, wave = tid >> 5;
  const int m0 = blockIdx.x * D;

  if (tid < 32) {
    tdm_load_tile_2d((unsigned int)(uintptr_t)sXc, agg + (size_t)m0 * D, D, D, D);
    tdm_load_tile_2d((unsigned int)(uintptr_t)sXr, hin + (size_t)m0 * D, D, D, D);
    tdm_load_tile_2d((unsigned int)(uintptr_t)sWc, wtc, D, D, D);
    tdm_load_tile_2d((unsigned int)(uintptr_t)sWr, wtr, D, D, D);
    __builtin_amdgcn_s_wait_tensorcnt(0);
  }
  __syncthreads();

  const int row = lane & 15, half = lane >> 4;
  const int kbA = half * 16, kbB = half * 16;
  const unsigned short* aC = sXc + (wave * 16 + row) * D;
  const unsigned short* aR = sXr + (wave * 16 + row) * D;

  v16bf afC[4], afR[4];
#pragma unroll
  for (int kk = 0; kk < 4; ++kk) {
    afC[kk] = load_afrag(aC + kk * 32 + kbA);
    afR[kk] = load_afrag(aR + kk * 32 + kbA);
  }

#pragma unroll
  for (int nt = 0; nt < 8; ++nt) {
    v8f accC, accR;
#pragma unroll
    for (int r = 0; r < 8; ++r) { accC[r] = 0.0f; accR[r] = 0.0f; }
    const unsigned short* bC = sWc + (nt * 16 + row) * D;
    const unsigned short* bR = sWr + (nt * 16 + row) * D;
#pragma unroll
    for (int kk = 0; kk < 4; ++kk) {
      accC = wmma_bf16(afC[kk], load_bfrag(bC + kk * 32 + kbB), accC);
      accR = wmma_bf16(afR[kk], load_bfrag(bR + kk * 32 + kbB), accR);
    }
    const int col = nt * 16 + row;
    const float biasC = bcl[col];
    const float biasR = brl[col];
    const int rbase = m0 + wave * 16 + half * 8;
#pragma unroll
    for (int r = 0; r < 8; ++r) {
      float c = accC[r] + biasC; c = c > 0.0f ? c : 0.0f;
      float s = accR[r] + biasR; s = s > 0.0f ? s : 0.0f;
      float o = c + s;
      const size_t idx = (size_t)(rbase + r) * D + col;
      if (last) fout[idx] = o;
      else      hout[idx] = f32_to_bf16(o);
    }
  }
}

// ---------------------------------------------------------------------------
extern "C" void kernel_launch(void* const* d_in, const int* in_sizes, int n_in,
                              void* d_out, int out_size, void* d_ws, size_t ws_size,
                              hipStream_t stream) {
  const float* nf  = (const float*)d_in[0];
  const int*   src = (const int*)d_in[1];
  const int*   dst = (const int*)d_in[2];
  const float* Wi  = (const float*)d_in[3];
  const float* Wc  = (const float*)d_in[4];
  const float* bc  = (const float*)d_in[5];
  const float* Wr  = (const float*)d_in[6];
  const float* br  = (const float*)d_in[7];
  float* out = (float*)d_out;

  char* ws = (char*)d_ws;
  const size_t HBYTES = (size_t)N_NODES * D * sizeof(unsigned short); // 128 MB
  unsigned short* hA  = (unsigned short*)(ws);
  unsigned short* hB  = (unsigned short*)(ws + HBYTES);
  unsigned short* agg = (unsigned short*)(ws + 2 * HBYTES);
  unsigned short* wt  = (unsigned short*)(ws + 3 * HBYTES);  // 6 x [128][128]
  unsigned short* wti = wt + 6 * D * D;                      //     [128][96]

  (void)hipFuncSetAttribute((const void*)gcn_layer_kernel,
                            hipFuncAttributeMaxDynamicSharedMemorySize,
                            4 * D * D * 2);

  gcn_prep_kernel<<<(6 * D * D + D * KPAD + 255) / 256, 256, 0, stream>>>(
      Wc, Wr, Wi, wt, wti);
  gcn_init_kernel<<<N_NODES / D, 256, 0, stream>>>(nf, wti, hA);

  unsigned short* cur = hA;
  unsigned short* nxt = hB;
  for (int l = 0; l < 3; ++l) {
    gcn_zero_kernel<<<(N_NODES * D / 8) / 256, 256, 0, stream>>>((uint4*)agg);
    gcn_scatter_kernel<<<(N_EDGES * 16) / 256, 256, 0, stream>>>(src, dst, cur, agg);
    gcn_layer_kernel<<<N_NODES / D, 256, 4 * D * D * 2, stream>>>(
        agg, cur, wt + l * D * D, wt + (3 + l) * D * D,
        bc + l * D, br + l * D, nxt, out, (l == 2) ? 1 : 0);
    unsigned short* t = cur; cur = nxt; nxt = t;
  }
}
